// RegionProposalNetwork_19215683683079
// MI455X (gfx1250) — compile-verified, matrix operand order checked by reading, not run on hardware
//
#include <hip/hip_runtime.h>
#include <hip/hip_bf16.h>
#include <stdint.h>

typedef __attribute__((ext_vector_type(16))) __bf16 bf16x16;
typedef __attribute__((ext_vector_type(8)))  float  f32x8;

#define C_IN   512
#define HW     64
#define NPIX   4096      // 64*64
#define KDIM   4608      // 512*9
#define NANCH  9
#define NBOX   36864     // NPIX*NANCH
#define NPRE   6000
#define NPOST  300
#define NSORT  65536
#define NMASKW 188       // ceil(6000/32)

// d_out layout (float elements)
#define LOC_OFF   0
#define SCORE_OFF 294912
#define ROI_OFF   442368
#define IDX_OFF   444768
#define ANCH_OFF  445368

// d_ws layout (bytes), all 128-aligned
#define WS_H     0ull          // bf16 h: 2*512*4096*2          = 8388608
#define WS_WT2   8388608ull    // bf16 wt2: 512*4608*2          = 4718592
#define WS_WC    13107200ull   // bf16 wcomb: 64*512*2          = 65536
#define WS_FG    13172736ull   // f32 fg: 2*36864*4             = 294912
#define WS_BOX   13467648ull   // f32 boxes: 2*36864*4*4        = 1179648
#define WS_KEY   14647296ull   // u64 keys: 2*65536*8           = 1048576
#define WS_TOPB  15695872ull   // f32 top boxes: 2*6000*4*4     = 192000
#define WS_TOPV  15887872ull   // u32 top valid: 2*6000*4       = 48000
#define WS_MASK  15935872ull   // u32 masks: 2*6000*188*4       = 9024000

// -------------------- Kernel 0: one-time weight transform to bf16 -----------
// wt2[m][tap*512+ci] = w1[m][ci*9+tap]   (tap-major K so the GEMM loop has no /9)
// wcomb[64][512]: rows 0-35 loc, 36-53 score, 54-63 zero.
__global__ __launch_bounds__(256)
void prep_weights(const float* __restrict__ w1, const float* __restrict__ wl,
                  const float* __restrict__ wsc,
                  __bf16* __restrict__ wt2, __bf16* __restrict__ wcomb)
{
  size_t i = (size_t)blockIdx.x * 256 + threadIdx.x;
  if (i < (size_t)512 * KDIM) {
    int m = (int)(i / KDIM);
    int r = (int)(i - (size_t)m * KDIM);
    int tap = r >> 9;        // r / 512
    int ci  = r & 511;
    wt2[i] = (__bf16)w1[(size_t)m * KDIM + ci * 9 + tap];
  }
  if (i < 64 * 512) {
    int m = (int)(i >> 9), k = (int)(i & 511);
    float v = 0.0f;
    if (m < 36)      v = wl[m * 512 + k];
    else if (m < 54) v = wsc[(m - 36) * 512 + k];
    wcomb[i] = (__bf16)v;
  }
}

// -------------------- Kernel A: 3x3 conv as implicit GEMM, bf16 WMMA --------
// grid: 2 batches * 8 Mtiles * 32 Ntiles = 512 blocks of 256 threads.
// Macro tile 64(M) x 128(N); 8 waves as 4(M)x2(N); wave = 16x64 = 4 WMMA accums.
// K loop: tap-major (9 taps x 16 channel-blocks of 32).
__global__ __launch_bounds__(256)
void conv3x3_wmma(const float* __restrict__ x, const __bf16* __restrict__ wt2,
                  const float* __restrict__ bias, __bf16* __restrict__ hout)
{
  int blk = blockIdx.x;
  int nt = blk & 31;
  int mt = (blk >> 5) & 7;
  int n  = blk >> 8;
  int m0 = mt * 64;
  int p0 = nt * 128;

  __shared__ __align__(16) __bf16 As[64][40];    // [m][k], 80B rows (16B aligned)
  __shared__ __align__(16) __bf16 Bs[128][40];   // [n][k]

  int tid  = threadIdx.x;
  int lane = tid & 31;
  int wave = tid >> 5;
  int wm = wave >> 1;
  int wn = wave & 1;
  int hi = lane >> 4;
  int lm = lane & 15;

  // staging coordinates (loop invariant)
  int a_m  = tid >> 2;          // 0..63
  int a_kb = (tid & 3) * 8;     // 0,8,16,24
  int b_nl = tid & 127;         // this thread's pixel column in the tile
  int b_kb = (tid >> 7) * 16;   // 0 or 16; thread owns kl = b_kb..b_kb+15
  int p    = p0 + b_nl;
  int py   = p >> 6, px = p & 63;
  const float* xn = x + (size_t)n * C_IN * NPIX;

  f32x8 acc[4] = {};

  for (int tap = 0; tap < 9; ++tap) {
    int dy = tap / 3 - 1, dx = tap - (tap / 3) * 3 - 1;   // scalar, 9x total
    bool bvalid = ((unsigned)(py + dy) < 64u) && ((unsigned)(px + dx) < 64u);
    // unconditional loads from a clamped (always in-bounds) address; zero via mask
    const float* bsrc = xn + (bvalid ? (p + dy * 64 + dx) : p);
    float bmask = bvalid ? 1.0f : 0.0f;
    const __bf16* asrc = wt2 + (size_t)(m0 + a_m) * KDIM + tap * 512 + a_kb;

    for (int cb = 0; cb < C_IN; cb += 32) {
      // A tile: one b128 global load + one b128 LDS store per thread
      *(uint4*)&As[a_m][a_kb] = *(const uint4*)(asrc + cb);

      // B tile: 16 coalesced, unconditional b32 loads issued back-to-back
      float bv[16];
#pragma unroll
      for (int i = 0; i < 16; ++i)
        bv[i] = bsrc[(size_t)(cb + b_kb + i) * NPIX];
      // mask + convert, then 2x b128 LDS stores (contiguous k-run in Bs[n][k])
      union { __bf16 e[16]; uint4 q[2]; } pk;
#pragma unroll
      for (int i = 0; i < 16; ++i)
        pk.e[i] = (__bf16)(bv[i] * bmask);
      *(uint4*)&Bs[b_nl][b_kb]     = pk.q[0];
      *(uint4*)&Bs[b_nl][b_kb + 8] = pk.q[1];
      __syncthreads();

      // fragments: per lane two contiguous 8-element runs -> 2x ds_load_b128
      union { bf16x16 v; uint4 q[2]; } fa, fb;
      fa.q[0] = *(const uint4*)&As[wm * 16 + lm][8 * hi];
      fa.q[1] = *(const uint4*)&As[wm * 16 + lm][16 + 8 * hi];
#pragma unroll
      for (int t = 0; t < 4; ++t) {
        int nn = wn * 64 + t * 16 + lm;
        fb.q[0] = *(const uint4*)&Bs[nn][8 * hi];
        fb.q[1] = *(const uint4*)&Bs[nn][16 + 8 * hi];
        acc[t] = __builtin_amdgcn_wmma_f32_16x16x32_bf16(
            false, fa.v, false, fb.v, (short)0, acc[t], false, false);
      }
      __syncthreads();
    }
  }

  // bias + relu, store h as bf16 [n][co][p]
#pragma unroll
  for (int t = 0; t < 4; ++t) {
#pragma unroll
    for (int r = 0; r < 8; ++r) {
      int co = m0 + wm * 16 + r + hi * 8;
      int pp = p0 + wn * 64 + t * 16 + lm;
      float v = acc[t][r] + bias[co];
      v = v > 0.0f ? v : 0.0f;
      hout[((size_t)n * C_IN + co) * NPIX + pp] = (__bf16)v;
    }
  }
}

// -------------------- Kernel B: fused 1x1 loc(36)+score(18) GEMM ------------
// M=64 (rows 0-35 loc, 36-53 score), K=512, N=8192; grid 64 blocks.
__global__ __launch_bounds__(256)
void conv1x1_wmma(const __bf16* __restrict__ hsrc, const __bf16* __restrict__ wcomb,
                  const float* __restrict__ lb, const float* __restrict__ sb,
                  float* __restrict__ out)
{
  int blk = blockIdx.x;
  int nt = blk & 31;
  int n  = blk >> 5;
  int p0 = nt * 128;

  __shared__ __align__(16) __bf16 As[64][40];
  __shared__ __align__(16) __bf16 Bs[128][40];

  int tid  = threadIdx.x;
  int lane = tid & 31;
  int wave = tid >> 5;
  int wm = wave >> 1;
  int wn = wave & 1;
  int hi = lane >> 4;
  int lm = lane & 15;

  int a_m  = tid >> 2;
  int a_kb = (tid & 3) * 8;

  f32x8 acc[4] = {};

  for (int cb = 0; cb < C_IN; cb += 32) {
    // A tile: one b128 load + store per thread (bf16 weights, m-major)
    *(uint4*)&As[a_m][a_kb] = *(const uint4*)&wcomb[(size_t)a_m * C_IN + cb + a_kb];
    // B tile: 2x b128 loads of h (contiguous in p), transpose-store to [n][k]
#pragma unroll
    for (int i = 0; i < 2; ++i) {
      int c   = tid + 256 * i;       // 0..511
      int kl  = c >> 4;              // 0..31
      int nl8 = (c & 15) * 8;        // 0..120
      union { uint4 q; __bf16 e[8]; } u;
      u.q = *(const uint4*)&hsrc[((size_t)n * C_IN + cb + kl) * NPIX + p0 + nl8];
#pragma unroll
      for (int j = 0; j < 8; ++j) Bs[nl8 + j][kl] = u.e[j];
    }
    __syncthreads();

    union { bf16x16 v; uint4 q[2]; } fa, fb;
    fa.q[0] = *(const uint4*)&As[wm * 16 + lm][8 * hi];
    fa.q[1] = *(const uint4*)&As[wm * 16 + lm][16 + 8 * hi];
#pragma unroll
    for (int t = 0; t < 4; ++t) {
      int nn = wn * 64 + t * 16 + lm;
      fb.q[0] = *(const uint4*)&Bs[nn][8 * hi];
      fb.q[1] = *(const uint4*)&Bs[nn][16 + 8 * hi];
      acc[t] = __builtin_amdgcn_wmma_f32_16x16x32_bf16(
          false, fa.v, false, fb.v, (short)0, acc[t], false, false);
    }
    __syncthreads();
  }

  // scatter into transposed rpn_locs / rpn_scores layout
#pragma unroll
  for (int t = 0; t < 4; ++t) {
#pragma unroll
    for (int r = 0; r < 8; ++r) {
      int ch = wm * 16 + r + hi * 8;
      int p  = p0 + wn * 64 + t * 16 + lm;
      float v = acc[t][r];
      if (ch < 36) {
        int a = ch >> 2, j = ch & 3;
        out[LOC_OFF + ((size_t)n * NBOX + (size_t)p * NANCH + a) * 4 + j] = v + lb[ch];
      } else if (ch < 54) {
        int c = ch - 36;
        int a = c >> 1, c2 = c & 1;
        out[SCORE_OFF + ((size_t)n * NBOX + (size_t)p * NANCH + a) * 2 + c2] = v + sb[c];
      }
    }
  }
}

// -------------------- Kernel C: fg scores, anchors, roi_indices, zero rois --
__global__ __launch_bounds__(256)
void rpn_misc(float* __restrict__ out, float* __restrict__ fg)
{
  int i = blockIdx.x * 256 + threadIdx.x;
  if (i < 2 * NBOX) {
    float s0 = out[SCORE_OFF + (size_t)i * 2 + 0];
    float s1 = out[SCORE_OFF + (size_t)i * 2 + 1];
    fg[i] = 1.0f / (1.0f + __expf(s0 - s1));   // softmax[...,1] of 2 logits
  }
  if (i < NBOX) {
    int p = i / NANCH, a = i - (i / NANCH) * NANCH;
    int ridx = a / 3, sidx = a - ridx * 3;
    float r = 0.5f * (float)(1 << ridx);       // 0.5, 1, 2
    float s = 8.0f * (float)(1 << sidx);       // 8, 16, 32
    float hh = 16.0f * s * sqrtf(r);
    float ww = 16.0f * s * sqrtf(1.0f / r);
    float cy = (float)(p >> 6) * 16.0f + 8.0f;
    float cx = (float)(p & 63) * 16.0f + 8.0f;
    out[ANCH_OFF + (size_t)i * 4 + 0] = cy - 0.5f * hh;
    out[ANCH_OFF + (size_t)i * 4 + 1] = cx - 0.5f * ww;
    out[ANCH_OFF + (size_t)i * 4 + 2] = cy + 0.5f * hh;
    out[ANCH_OFF + (size_t)i * 4 + 3] = cx + 0.5f * ww;
  }
  if (i < 2 * NPOST) ((int*)out)[IDX_OFF + i] = i / NPOST;
  if (i < 2 * NPOST * 4) out[ROI_OFF + i] = 0.0f;
}

// -------------------- Kernel D: loc2bbox + clip + minsize + sort keys -------
__global__ __launch_bounds__(256)
void rpn_boxes(const float* __restrict__ out, const float* __restrict__ fg,
               float* __restrict__ boxes, unsigned long long* __restrict__ keys,
               const int* __restrict__ imgh, const int* __restrict__ imgw)
{
  int t = blockIdx.x * 256 + threadIdx.x;     // < 2*65536
  if (t >= 2 * NSORT) return;
  int n = t >> 16, s = t & (NSORT - 1);
  unsigned long long key;
  if (s < NBOX) {
    const float* A = out + ANCH_OFF + (size_t)s * 4;
    const float* L = out + LOC_OFF + ((size_t)n * NBOX + s) * 4;
    float ah = A[2] - A[0], aw = A[3] - A[1];
    float acy = A[0] + 0.5f * ah, acx = A[1] + 0.5f * aw;
    float cy = L[0] * ah + acy, cx = L[1] * aw + acx;
    float bh = __expf(L[2]) * ah, bw = __expf(L[3]) * aw;
    float H = (float)imgh[0], W = (float)imgw[0];
    float y1 = fminf(fmaxf(cy - 0.5f * bh, 0.0f), H);
    float x1 = fminf(fmaxf(cx - 0.5f * bw, 0.0f), W);
    float y2 = fminf(fmaxf(cy + 0.5f * bh, 0.0f), H);
    float x2 = fminf(fmaxf(cx + 0.5f * bw, 0.0f), W);
    float* B = boxes + ((size_t)n * NBOX + s) * 4;
    B[0] = y1; B[1] = x1; B[2] = y2; B[3] = x2;
    bool valid = (y2 - y1 >= 16.0f) && (x2 - x1 >= 16.0f);
    unsigned ms;
    if (valid) {
      unsigned fb = __float_as_uint(fg[n * NBOX + s]);
      ms = (fb & 0x80000000u) ? ~fb : (fb | 0x80000000u);
    } else {
      ms = 0x007FFFFFu;  // mapped -inf
    }
    key = ((unsigned long long)ms << 32) | (unsigned)(0xFFFFFFFFu - (unsigned)s);
  } else {
    key = (unsigned long long)(0xFFFFFFFFu - (unsigned)s);  // pad, sorts last
  }
  keys[(size_t)n * NSORT + s] = key;
}

// -------------------- Kernel E: one bitonic pass (descending) ---------------
__global__ __launch_bounds__(256)
void bitonic_pass(unsigned long long* __restrict__ keys, int k, int j)
{
  int t = blockIdx.x * 256 + threadIdx.x;     // 2*65536 threads
  if (t >= 2 * NSORT) return;
  int n = t >> 16, i = t & (NSORT - 1);
  int l = i ^ j;
  if (l > i) {
    unsigned long long* kk = keys + (size_t)n * NSORT;
    unsigned long long a = kk[i], b = kk[l];
    bool desc = ((i & k) == 0);
    if (desc ? (a < b) : (a > b)) { kk[i] = b; kk[l] = a; }
  }
}

// -------------------- Kernel F: gather top-6000 -----------------------------
__global__ __launch_bounds__(256)
void rpn_gather(const unsigned long long* __restrict__ keys,
                const float* __restrict__ boxes,
                float* __restrict__ topb, unsigned* __restrict__ topv)
{
  int t = blockIdx.x * 256 + threadIdx.x;
  if (t >= 2 * NPRE) return;
  int n = t / NPRE, r = t - n * NPRE;
  unsigned long long key = keys[(size_t)n * NSORT + r];
  unsigned hiK = (unsigned)(key >> 32);
  unsigned valid = (hiK >= 0x80000000u) ? 1u : 0u;
  float b0 = 0, b1 = 0, b2 = 0, b3 = 0;
  if (valid) {
    unsigned idx = 0xFFFFFFFFu - (unsigned)key;
    const float* B = boxes + ((size_t)n * NBOX + idx) * 4;
    b0 = B[0]; b1 = B[1]; b2 = B[2]; b3 = B[3];
  }
  float* T = topb + ((size_t)n * NPRE + r) * 4;
  T[0] = b0; T[1] = b1; T[2] = b2; T[3] = b3;
  topv[(size_t)n * NPRE + r] = valid;
}

// -------------------- Kernel G: IoU suppression bitmasks --------------------
__global__ __launch_bounds__(256)
void iou_mask(const float* __restrict__ topb, unsigned* __restrict__ mask)
{
  int t = blockIdx.x * 256 + threadIdx.x;
  if (t >= 2 * NPRE * NMASKW) return;
  int n = t / (NPRE * NMASKW);
  int rem = t - n * (NPRE * NMASKW);
  int row = rem / NMASKW;
  int jb  = rem - row * NMASKW;
  const float* bi = topb + ((size_t)n * NPRE + row) * 4;
  float iy1 = bi[0], ix1 = bi[1], iy2 = bi[2], ix2 = bi[3];
  float areai = (iy2 - iy1) * (ix2 - ix1);
  unsigned bits = 0;
  int jbase = jb * 32;
#pragma unroll 4
  for (int j = 0; j < 32; ++j) {
    int jg = jbase + j;
    if (jg < NPRE && jg > row) {
      const float* bj = topb + ((size_t)n * NPRE + jg) * 4;
      float ty = fmaxf(iy1, bj[0]), tx = fmaxf(ix1, bj[1]);
      float by = fminf(iy2, bj[2]), bx = fminf(ix2, bj[3]);
      float inter = fmaxf(by - ty, 0.0f) * fmaxf(bx - tx, 0.0f);
      float areaj = (bj[2] - bj[0]) * (bj[3] - bj[1]);
      float iou = inter / (areai + areaj - inter + 1e-9f);
      if (iou > 0.7f) bits |= (1u << j);
    }
  }
  mask[((size_t)n * NPRE + row) * NMASKW + jb] = bits;
}

// -------------------- Kernel H: sequential greedy NMS scan ------------------
__global__ __launch_bounds__(32)
void nms_scan(const float* __restrict__ topb, const unsigned* __restrict__ topv,
              const unsigned* __restrict__ mask, float* __restrict__ out)
{
  int n = blockIdx.x;
  int lane = threadIdx.x;
  __shared__ unsigned rem[NMASKW];
  __shared__ int cnt;
  for (int w = lane; w < NMASKW; w += 32) rem[w] = 0;
  if (lane == 0) cnt = 0;
  __syncthreads();
  for (int i = 0; i < NPRE; ++i) {
    __syncthreads();
    if (cnt >= NPOST) break;
    bool kept = (topv[(size_t)n * NPRE + i] != 0) &&
                !((rem[i >> 5] >> (i & 31)) & 1u);
    if (kept) {
      const unsigned* mrow = mask + ((size_t)n * NPRE + i) * NMASKW;
      for (int w = lane; w < NMASKW; w += 32) rem[w] |= mrow[w];
      if (lane == 0) {
        int c = cnt;
        const float* B = topb + ((size_t)n * NPRE + i) * 4;
        size_t o = ROI_OFF + ((size_t)n * NPOST + c) * 4;
        out[o + 0] = B[0]; out[o + 1] = B[1];
        out[o + 2] = B[2]; out[o + 3] = B[3];
        cnt = c + 1;
      }
    }
  }
}

// ----------------------------------------------------------------------------
extern "C" void kernel_launch(void* const* d_in, const int* in_sizes, int n_in,
                              void* d_out, int out_size, void* d_ws, size_t ws_size,
                              hipStream_t stream)
{
  const float* x    = (const float*)d_in[0];   // feature_map
  const float* w1   = (const float*)d_in[1];   // conv1_w
  const float* b1   = (const float*)d_in[2];   // conv1_b
  const float* sw   = (const float*)d_in[3];   // score_w
  const float* sb   = (const float*)d_in[4];   // score_b
  const float* lw   = (const float*)d_in[5];   // loc_w
  const float* lb   = (const float*)d_in[6];   // loc_b
  const int*   imgh = (const int*)d_in[7];
  const int*   imgw = (const int*)d_in[8];
  float* out = (float*)d_out;

  char* ws = (char*)d_ws;
  __bf16*             hbuf  = (__bf16*)(ws + WS_H);
  __bf16*             wt2   = (__bf16*)(ws + WS_WT2);
  __bf16*             wcomb = (__bf16*)(ws + WS_WC);
  float*              fgbuf = (float*)(ws + WS_FG);
  float*              boxes = (float*)(ws + WS_BOX);
  unsigned long long* keys  = (unsigned long long*)(ws + WS_KEY);
  float*              topb  = (float*)(ws + WS_TOPB);
  unsigned*           topv  = (unsigned*)(ws + WS_TOPV);
  unsigned*           maskp = (unsigned*)(ws + WS_MASK);

  prep_weights<<<(512 * KDIM + 255) / 256, 256, 0, stream>>>(w1, lw, sw, wt2, wcomb);
  conv3x3_wmma<<<512, 256, 0, stream>>>(x, wt2, b1, hbuf);
  conv1x1_wmma<<<64, 256, 0, stream>>>(hbuf, wcomb, lb, sb, out);
  rpn_misc<<<(2 * NBOX + 255) / 256, 256, 0, stream>>>(out, fgbuf);
  rpn_boxes<<<(2 * NSORT + 255) / 256, 256, 0, stream>>>(out, fgbuf, boxes, keys,
                                                         imgh, imgw);
  for (int k = 2; k <= NSORT; k <<= 1)
    for (int j = k >> 1; j > 0; j >>= 1)
      bitonic_pass<<<(2 * NSORT + 255) / 256, 256, 0, stream>>>(keys, k, j);
  rpn_gather<<<(2 * NPRE + 255) / 256, 256, 0, stream>>>(keys, boxes, topb, topv);
  iou_mask<<<(2 * NPRE * NMASKW + 255) / 256, 256, 0, stream>>>(topb, maskp);
  nms_scan<<<2, 32, 0, stream>>>(topb, topv, maskp, out);
}
